// PointGroup_83519934038505
// MI455X (gfx1250) — matrix operand order repaired and checked by reading, not compiled, not run on hardware
//
#include <hip/hip_runtime.h>

typedef __attribute__((ext_vector_type(16))) _Float16 v16h;
typedef __attribute__((ext_vector_type(8)))  float    v8f;

#define MAX_CT 7
#define BN_EPS 1e-4f

// ---------------------------------------------------------------------------
// Weight packing: raw f32 [Cout, Cin, taps] -> f16 WMMA A-fragment order.
// A order used here (16x32 f16, contiguous k): lane = m + 16*(k/16), elem e = k%16.
// Inverse used here: given (lane, e): m = lane&15, hi = lane>>4,
//   k = hi*16 + e.
// K ordering: k = tap*Cin + ci (tapSel<0) or k = ci with fixed tap (tapSel>=0).
// ---------------------------------------------------------------------------
__global__ void pack_w_k(const float* __restrict__ w, _Float16* __restrict__ dst,
                         int Cout, int Cin, int kk3, int tapSel, int Ktot, int nK) {
    int t = blockIdx.x * blockDim.x + threadIdx.x;
    int total = (Cout / 16) * nK * 32;
    if (t >= total) return;
    int lane = t & 31;
    int rest = t >> 5;
    int ks = rest % nK;
    int ct = rest / nK;
    int m  = ct * 16 + (lane & 15);
    int hi = lane >> 4;
    v16h vals;
#pragma unroll
    for (int e = 0; e < 16; ++e) {
        int k = ks * 32 + hi * 16 + e;
        float v = 0.f;
        if (k < Ktot) {
            int tap, ci;
            if (tapSel >= 0) { tap = tapSel; ci = k; }
            else             { tap = k / Cin; ci = k - tap * Cin; }
            v = w[((size_t)m * Cin + ci) * kk3 + tap];
        }
        vals[e] = (_Float16)v;
    }
    *(v16h*)(dst + (size_t)t * 16) = vals;
}

// ---------------------------------------------------------------------------
// Implicit-GEMM conv (cubic dims). One wave per block: 32 output voxels
// (two 16-wide N tiles) x all Cout tiles. Each A fragment load feeds two
// WMMAs. B fragments built per K-step with incremental k->(ci,kd,kh,kw)
// (no per-element divisions); the k<Ktot guard runs only in the tail step.
// ---------------------------------------------------------------------------
__global__ __launch_bounds__(32)
void conv_wmma_k(const _Float16* __restrict__ act, const _Float16* __restrict__ pw,
                 float* __restrict__ out, int Cin, int Sin, int Cout, int Sout,
                 int kk, int stride, int pad, int Ktot, int nK, int accflag) {
    int lane  = threadIdx.x;
    int n     = lane & 15;
    int khalf = lane >> 4;
    int nvox  = Sout * Sout * Sout;
    int v0    = blockIdx.x * 32 + n;
    int v1    = v0 + 16;
    bool ok0  = v0 < nvox;
    bool ok1  = v1 < nvox;

    int ctn  = Cout >> 4;
    int kk2  = kk * kk;
    int Sin2 = Sin * Sin;
    int Sin3 = Sin2 * Sin;

    // spatial receptive-field origins for both N tiles
    int idA = 0, ihA = 0, iwA = 0, idB = 0, ihB = 0, iwB = 0;
    if (ok0) {
        int od = v0 / (Sout * Sout);
        int r  = v0 - od * Sout * Sout;
        int oh = r / Sout;
        int ow = r - oh * Sout;
        idA = od * stride - pad; ihA = oh * stride - pad; iwA = ow * stride - pad;
    }
    if (ok1) {
        int od = v1 / (Sout * Sout);
        int r  = v1 - od * Sout * Sout;
        int oh = r / Sout;
        int ow = r - oh * Sout;
        idB = od * stride - pad; ihB = oh * stride - pad; iwB = ow * stride - pad;
    }

    v8f acc[MAX_CT][2];
#pragma unroll
    for (int ct = 0; ct < MAX_CT; ++ct) {
#pragma unroll
        for (int e = 0; e < 8; ++e) {
            float c0 = 0.f, c1 = 0.f;
            if (accflag && ct < ctn) {
                size_t ch = (size_t)(ct * 16 + khalf * 8 + e) * nvox;
                if (ok0) c0 = out[ch + v0];
                if (ok1) c1 = out[ch + v1];
            }
            acc[ct][0][e] = c0;
            acc[ct][1][e] = c1;
        }
    }

    int nKmain = Ktot >> 5;  // K-steps with no k<Ktot masking needed

    auto kstep = [&](int ks, bool tail) {
        int kbase = ks * 32 + khalf * 16;
        // decompose once per K-step; update incrementally per element
        int tap = kbase / Cin;
        int ci  = kbase - tap * Cin;
        int kd  = tap / kk2;
        int rr  = tap - kd * kk2;
        int kh  = rr / kk;
        int kw  = rr - kh * kk;

        v16h bfragA, bfragB;
#pragma unroll
        for (int e = 0; e < 16; ++e) {
            bool kok = !tail || (kbase + e) < Ktot;
            size_t cbase = (size_t)ci * Sin3;
            _Float16 va = (_Float16)0.f, vb = (_Float16)0.f;
            {
                int id = idA + kd, ih = ihA + kh, iw = iwA + kw;
                if (ok0 && kok && (unsigned)id < (unsigned)Sin &&
                    (unsigned)ih < (unsigned)Sin && (unsigned)iw < (unsigned)Sin)
                    va = act[cbase + id * Sin2 + ih * Sin + iw];
            }
            {
                int id = idB + kd, ih = ihB + kh, iw = iwB + kw;
                if (ok1 && kok && (unsigned)id < (unsigned)Sin &&
                    (unsigned)ih < (unsigned)Sin && (unsigned)iw < (unsigned)Sin)
                    vb = act[cbase + id * Sin2 + ih * Sin + iw];
            }
            bfragA[e] = va;
            bfragB[e] = vb;
            // k -> k+1 : ci++ with carry into (kw,kh,kd)
            ci++;
            if (ci == Cin) {
                ci = 0;
                kw++;
                if (kw == kk) {
                    kw = 0;
                    kh++;
                    if (kh == kk) { kh = 0; kd++; }
                }
            }
        }

        if (ks + 1 < nK)  // keep the sequential A stream warm (global_prefetch_b8)
            __builtin_prefetch(pw + (((size_t)ks + 1) * 32 + lane) * 16, 0, 1);

#pragma unroll
        for (int ct = 0; ct < MAX_CT; ++ct) {
            if (ct < ctn) {
                v16h afrag = *(const v16h*)(pw + (((size_t)ct * nK + ks) * 32 + lane) * 16);
                acc[ct][0] = __builtin_amdgcn_wmma_f32_16x16x32_f16(
                    false, afrag, false, bfragA, (short)0, acc[ct][0], false, false);
                acc[ct][1] = __builtin_amdgcn_wmma_f32_16x16x32_f16(
                    false, afrag, false, bfragB, (short)0, acc[ct][1], false, false);
            }
        }
    };

    for (int ks = 0; ks < nKmain; ++ks) kstep(ks, false);
    for (int ks = nKmain; ks < nK; ++ks) kstep(ks, true);

#pragma unroll
    for (int ct = 0; ct < MAX_CT; ++ct) {
        if (ct < ctn) {
#pragma unroll
            for (int e = 0; e < 8; ++e) {
                size_t ch = (size_t)(ct * 16 + khalf * 8 + e) * nvox;
                if (ok0) out[ch + v0] = acc[ct][0][e];
                if (ok1) out[ch + v1] = acc[ct][1][e];
            }
        }
    }
}

// ---------------------------------------------------------------------------
// Transposed conv k2 s2 as 8 parity-octant 1x1 GEMMs with strided scatter into
// the concat buffer (channel offset coutOff, fine edge Sf = 2*Sc).
// ---------------------------------------------------------------------------
__global__ __launch_bounds__(32)
void up_wmma_k(const _Float16* __restrict__ act, const _Float16* __restrict__ pw,
               float* __restrict__ out, int Cin, int Sc, int Cout, int coutOff,
               int nK, int perOct) {
    int lane  = threadIdx.x;
    int oct   = blockIdx.y;
    int oz = (oct >> 2) & 1, oy = (oct >> 1) & 1, ox = oct & 1;
    int n     = lane & 15;
    int khalf = lane >> 4;
    int nvox  = Sc * Sc * Sc;
    int v     = blockIdx.x * 16 + n;
    bool vok  = v < nvox;
    int cz = 0, cy = 0, cx = 0;
    if (vok) {
        cz = v / (Sc * Sc);
        int r = v - cz * Sc * Sc;
        cy = r / Sc;
        cx = r - cy * Sc;
    }
    int Sf  = 2 * Sc;
    int ctn = Cout >> 4;
    const _Float16* pwo = pw + (size_t)oct * perOct;

    v8f acc[MAX_CT];
#pragma unroll
    for (int ct = 0; ct < MAX_CT; ++ct)
#pragma unroll
        for (int e = 0; e < 8; ++e) acc[ct][e] = 0.f;

    int nKmain = Cin >> 5;

    auto kstep = [&](int ks, bool tail) {
        int kbase = ks * 32 + khalf * 16;
        const _Float16* ap = act + (size_t)kbase * nvox + v;  // strides by nvox per k
        v16h bfrag;
#pragma unroll
        for (int e = 0; e < 16; ++e) {
            _Float16 val = (_Float16)0.f;
            if (vok && (!tail || (kbase + e) < Cin)) val = *ap;
            bfrag[e] = val;
            ap += nvox;
        }

        if (ks + 1 < nK)
            __builtin_prefetch(pwo + (((size_t)ks + 1) * 32 + lane) * 16, 0, 1);

#pragma unroll
        for (int ct = 0; ct < MAX_CT; ++ct) {
            if (ct < ctn) {
                v16h afrag = *(const v16h*)(pwo + (((size_t)ct * nK + ks) * 32 + lane) * 16);
                acc[ct] = __builtin_amdgcn_wmma_f32_16x16x32_f16(
                    false, afrag, false, bfrag, (short)0, acc[ct], false, false);
            }
        }
    };

    for (int ks = 0; ks < nKmain; ++ks) kstep(ks, false);
    for (int ks = nKmain; ks < nK; ++ks) kstep(ks, true);

    if (vok) {
        int fz = 2 * cz + oz, fy = 2 * cy + oy, fx = 2 * cx + ox;
        size_t sp = ((size_t)fz * Sf + fy) * Sf + fx;
        size_t chs = (size_t)Sf * Sf * Sf;
#pragma unroll
        for (int ct = 0; ct < MAX_CT; ++ct) {
            if (ct < ctn) {
#pragma unroll
                for (int e = 0; e < 8; ++e) {
                    int m = ct * 16 + khalf * 8 + e;
                    out[(size_t)(coutOff + m) * chs + sp] = acc[ct][e];
                }
            }
        }
    }
}

// ---------------------------------------------------------------------------
// BatchNorm statistics: one workgroup per channel, LDS tree reduction.
// mv[2c] = mean, mv[2c+1] = rsqrt(var + eps).
// ---------------------------------------------------------------------------
__global__ __launch_bounds__(256)
void bn_stats_k(const float* __restrict__ x, float* __restrict__ mv, int vol) {
    int c = blockIdx.x;
    const float* p = x + (size_t)c * vol;
    float s = 0.f, s2 = 0.f;
    for (int i = threadIdx.x; i < vol; i += 256) {
        float v = p[i];
        s += v;
        s2 += v * v;
    }
    __shared__ float sh1[256];
    __shared__ float sh2[256];
    sh1[threadIdx.x] = s;
    sh2[threadIdx.x] = s2;
    __syncthreads();
    for (int off = 128; off > 0; off >>= 1) {
        if ((int)threadIdx.x < off) {
            sh1[threadIdx.x] += sh1[threadIdx.x + off];
            sh2[threadIdx.x] += sh2[threadIdx.x + off];
        }
        __syncthreads();
    }
    if (threadIdx.x == 0) {
        float inv = 1.f / (float)vol;
        float m = sh1[0] * inv;
        float var = sh2[0] * inv - m * m;
        mv[2 * c]     = m;
        mv[2 * c + 1] = rsqrtf(var + BN_EPS);
    }
}

__global__ void bn_relu_f16_k(const float* __restrict__ x, const float* __restrict__ mv,
                              const float* __restrict__ g, const float* __restrict__ b,
                              _Float16* __restrict__ y, int vol, long n) {
    long i = (long)blockIdx.x * blockDim.x + threadIdx.x;
    if (i >= n) return;
    int c = (int)(i / vol);
    float v = (x[i] - mv[2 * c]) * mv[2 * c + 1] * g[c] + b[c];
    y[i] = (_Float16)fmaxf(v, 0.f);
}

__global__ void bn_relu_f32_k(const float* __restrict__ x, const float* __restrict__ mv,
                              const float* __restrict__ g, const float* __restrict__ b,
                              float* __restrict__ y, int vol, long n) {
    long i = (long)blockIdx.x * blockDim.x + threadIdx.x;
    if (i >= n) return;
    int c = (int)(i / vol);
    float v = (x[i] - mv[2 * c]) * mv[2 * c + 1] * g[c] + b[c];
    y[i] = fmaxf(v, 0.f);
}

__global__ void cvt_f16_k(const float* __restrict__ x, _Float16* __restrict__ y, long n) {
    long i = (long)blockIdx.x * blockDim.x + threadIdx.x;
    if (i < n) y[i] = (_Float16)x[i];
}

__global__ void copy_f32_k(const float* __restrict__ x, float* __restrict__ y, long n) {
    long i = (long)blockIdx.x * blockDim.x + threadIdx.x;
    if (i < n) y[i] = x[i];
}

// ---------------------------------------------------------------------------
// Host-side orchestration
// ---------------------------------------------------------------------------
namespace {

struct Bump {
    char*  base;
    size_t off;
    size_t cap;
    void* get(size_t n) {
        size_t a = (off + 255) & ~(size_t)255;
        off = a + n;
        return (void*)(base + a);
    }
};

struct BlkP { const float *g1, *b1, *w1, *g2, *b2, *w2, *iw; };
struct UP {
    BlkP blocks[2];
    const float *dg, *db, *dw, *ug, *ub, *uw;
    BlkP tail[2];
    bool has;
};

void read_blk(BlkP& b, bool has_iw, void* const* din, int& i) {
    b.g1 = (const float*)din[i++]; b.b1 = (const float*)din[i++];
    b.w1 = (const float*)din[i++]; b.g2 = (const float*)din[i++];
    b.b2 = (const float*)din[i++]; b.w2 = (const float*)din[i++];
    b.iw = has_iw ? (const float*)din[i++] : nullptr;
}

void read_ublock(UP* ns, int l, void* const* din, int& i) {
    UP& u = ns[l];
    read_blk(u.blocks[0], false, din, i);
    read_blk(u.blocks[1], false, din, i);
    u.has = (l < 6);
    if (u.has) {
        u.dg = (const float*)din[i++]; u.db = (const float*)din[i++];
        u.dw = (const float*)din[i++];
        read_ublock(ns, l + 1, din, i);
        u.ug = (const float*)din[i++]; u.ub = (const float*)din[i++];
        u.uw = (const float*)din[i++];
        read_blk(u.tail[0], true, din, i);   // cin = 2P != P -> i_branch
        read_blk(u.tail[1], false, din, i);
    }
}

inline int cdiv(long a, long b) { return (int)((a + b - 1) / b); }

void launch_conv(Bump& al, hipStream_t st, const _Float16* act, const float* wraw,
                 float* out, int Cin, int Sin, int Cout, int Sout,
                 int kk, int stride, int pad, int accflag) {
    int Ktot = Cin * kk * kk * kk;
    int nK   = (Ktot + 31) / 32;
    int ctn  = Cout / 16;
    _Float16* pw = (_Float16*)al.get((size_t)ctn * nK * 512 * 2);
    int tot = ctn * nK * 32;
    pack_w_k<<<cdiv(tot, 256), 256, 0, st>>>(wraw, pw, Cout, Cin, kk * kk * kk, -1, Ktot, nK);
    int nvox = Sout * Sout * Sout;
    conv_wmma_k<<<cdiv(nvox, 32), 32, 0, st>>>(act, pw, out, Cin, Sin, Cout, Sout,
                                               kk, stride, pad, Ktot, nK, accflag);
}

void launch_up(Bump& al, hipStream_t st, const _Float16* act, const float* wraw,
               float* outBase, int Cin, int Sc, int Cout, int coutOff) {
    int nK  = (Cin + 31) / 32;
    int ctn = Cout / 16;
    size_t per = (size_t)ctn * nK * 512;
    _Float16* pw = (_Float16*)al.get(per * 8 * 2);
    int tot = ctn * nK * 32;
    for (int o = 0; o < 8; ++o)
        pack_w_k<<<cdiv(tot, 256), 256, 0, st>>>(wraw, pw + per * o, Cout, Cin, 8, o, Cin, nK);
    int nvox = Sc * Sc * Sc;
    dim3 g(cdiv(nvox, 16), 8);
    up_wmma_k<<<g, 32, 0, st>>>(act, pw, outBase, Cin, Sc, Cout, coutOff, nK, (int)per);
}

_Float16* launch_bnrelu(Bump& al, hipStream_t st, const float* x, const float* g,
                        const float* b, int C, int vol) {
    float* mv = (float*)al.get((size_t)2 * C * 4);
    _Float16* y = (_Float16*)al.get((size_t)C * vol * 2);
    bn_stats_k<<<C, 256, 0, st>>>(x, mv, vol);
    long n = (long)C * vol;
    bn_relu_f16_k<<<cdiv(n, 256), 256, 0, st>>>(x, mv, g, b, y, vol, n);
    return y;
}

void run_block(Bump& al, hipStream_t st, const BlkP& bp, const float* x, float* out,
               int cin, int cout, int S) {
    int vol = S * S * S;
    // identity branch -> out
    if (bp.iw) {
        _Float16* xh = (_Float16*)al.get((size_t)cin * vol * 2);
        long n = (long)cin * vol;
        cvt_f16_k<<<cdiv(n, 256), 256, 0, st>>>(x, xh, n);
        launch_conv(al, st, xh, bp.iw, out, cin, S, cout, S, 1, 1, 0, 0);
    } else {
        long n = (long)cin * vol;
        copy_f32_k<<<cdiv(n, 256), 256, 0, st>>>(x, out, n);
    }
    // main branch, second conv accumulates into identity
    _Float16* a1 = launch_bnrelu(al, st, x, bp.g1, bp.b1, cin, vol);
    float* t1 = (float*)al.get((size_t)cout * vol * 4);
    launch_conv(al, st, a1, bp.w1, t1, cin, S, cout, S, 3, 1, 1, 0);
    _Float16* a2 = launch_bnrelu(al, st, t1, bp.g2, bp.b2, cout, vol);
    launch_conv(al, st, a2, bp.w2, out, cout, S, cout, S, 3, 1, 1, 1);
}

void run_ublock(Bump& al, hipStream_t st, UP* ns, int l, float* x) {
    int P = 16 * (l + 1);
    int S = 64 >> l;
    int vol = S * S * S;
    size_t mk0 = al.off;
    UP& u = ns[l];
    float* cur = x;
    for (int b = 0; b < 2; ++b) {
        float* out = (float*)al.get((size_t)P * vol * 4);
        size_t mk = al.off;
        run_block(al, st, u.blocks[b], cur, out, P, P, S);
        al.off = mk;
        cur = out;
    }
    if (u.has) {
        int P2 = 16 * (l + 2), S2 = S / 2, vol2 = S2 * S2 * S2;
        float* d = (float*)al.get((size_t)P2 * vol2 * 4);
        {
            size_t mk = al.off;
            _Float16* a = launch_bnrelu(al, st, cur, u.dg, u.db, P, vol);
            launch_conv(al, st, a, u.dw, d, P, S, P2, S2, 2, 2, 0, 0);
            al.off = mk;
        }
        run_ublock(al, st, ns, l + 1, d);
        float* cat = (float*)al.get((size_t)2 * P * vol * 4);
        {
            long n = (long)P * vol;
            copy_f32_k<<<cdiv(n, 256), 256, 0, st>>>(cur, cat, n);
            size_t mk = al.off;
            _Float16* a2 = launch_bnrelu(al, st, d, u.ug, u.ub, P2, vol2);
            launch_up(al, st, a2, u.uw, cat, P2, S2, P, P);
            al.off = mk;
        }
        cur = cat;
        for (int t = 0; t < 2; ++t) {
            int cin = (t == 0) ? 2 * P : P;
            float* out = (float*)al.get((size_t)P * vol * 4);
            size_t mk = al.off;
            run_block(al, st, u.tail[t], cur, out, cin, P, S);
            al.off = mk;
            cur = out;
        }
    }
    if (cur != x) {
        long n = (long)P * vol;
        copy_f32_k<<<cdiv(n, 256), 256, 0, st>>>(cur, x, n);
    }
    al.off = mk0;
}

} // namespace

extern "C" void kernel_launch(void* const* d_in, const int* in_sizes, int n_in,
                              void* d_out, int out_size, void* d_ws, size_t ws_size,
                              hipStream_t stream) {
    (void)in_sizes; (void)n_in; (void)out_size;
    int idx = 0;
    const float* x    = (const float*)d_in[idx++];
    const float* in_w = (const float*)d_in[idx++];
    static UP nodes[7];
    read_ublock(nodes, 0, d_in, idx);
    const float* out_g = (const float*)d_in[idx++];
    const float* out_b = (const float*)d_in[idx++];

    Bump al{(char*)d_ws, 0, ws_size};
    const int S0 = 64;
    const long vol0 = (long)S0 * S0 * S0;

    // input conv: x (6ch) -> h (16ch)
    _Float16* xh = (_Float16*)al.get((size_t)6 * vol0 * 2);
    long nx = 6 * vol0;
    cvt_f16_k<<<cdiv(nx, 256), 256, 0, stream>>>(x, xh, nx);
    float* h = (float*)al.get((size_t)16 * vol0 * 4);
    launch_conv(al, stream, xh, in_w, h, 6, S0, 16, S0, 3, 1, 1, 0);

    // U-Net recursion (in-place on h)
    run_ublock(al, stream, nodes, 0, h);

    // output layer: BN + ReLU -> d_out (f32)
    float* mv = (float*)al.get(2 * 16 * 4);
    bn_stats_k<<<16, 256, 0, stream>>>(h, mv, (int)vol0);
    long nout = 16 * vol0;
    bn_relu_f32_k<<<cdiv(nout, 256), 256, 0, stream>>>(h, mv, out_g, out_b,
                                                       (float*)d_out, (int)vol0, nout);
}
